// MultiHeadAttention_25967372271829
// MI455X (gfx1250) — matrix operand order repaired and checked by reading, not compiled, not run on hardware
//
#include <hip/hip_runtime.h>

typedef __attribute__((ext_vector_type(16))) _Float16 v16h;
typedef __attribute__((ext_vector_type(8)))  _Float16 v8h;
typedef __attribute__((ext_vector_type(4)))  _Float16 v4h;
typedef __attribute__((ext_vector_type(8)))  float    v8f;

#define WB 4096
#define LL 64
#define CC 256
#define HH 8
#define DD 32
#define MH 384

// LDS layout in halves (total 81920 halves = 160KB)
#define PS_OFF 0        // P matrices: 8 heads x 64 x 64  (aliases x-staging)
#define PS_STR 64
#define XS_STR 264      // x staging: 64 x 264 (padded) inside PS region
#define QS_OFF 32768    // q: 8 heads x 64 x 32
#define KS_OFF 49152    // k: 8 heads x 64 x 32
#define VT_OFF 65536    // v^T: 8 heads x 32 x 64
#define SM_HALVES 81920

__device__ __forceinline__ v8f wmma_f16(v16h a, v16h b, v8f c) {
  return __builtin_amdgcn_wmma_f32_16x16x32_f16(false, a, false, b, (short)0, c,
                                                false, false);
}

// A-operand 16x32 tile from row-major [row][K] storage.
// lanes 0-15: M=lane,   K = {kb..kb+7, kb+16..kb+23}, kb=0
// lanes16-31: M=lane-16, same with kb=8
__device__ __forceinline__ v16h tileA(const _Float16* base, int stride,
                                      int rowbase, int kbase, int lane) {
  const _Float16* p = base + (rowbase + (lane & 15)) * stride + kbase +
                      ((lane & 16) ? 8 : 0);
  v8h lo = *(const v8h*)p;
  v8h hi = *(const v8h*)(p + 16);
  return __builtin_shufflevector(lo, hi, 0, 1, 2, 3, 4, 5, 6, 7, 8, 9, 10, 11,
                                 12, 13, 14, 15);
}

// B-operand 32x16 tile from N-major [n][K] storage.
// lanes 0-15: N=lane, K = {kb..kb+7, kb+16..kb+23}, kb=0 ; lanes 16-31: N=lane-16, kb=8
__device__ __forceinline__ v16h tileB(const _Float16* base, int stride,
                                      int nbase, int kbase, int lane) {
  const _Float16* p = base + (nbase + (lane & 15)) * stride + kbase +
                      ((lane & 16) ? 8 : 0);
  v8h lo = *(const v8h*)p;
  v8h hi = *(const v8h*)(p + 16);
  return __builtin_shufflevector(lo, hi, 0, 1, 2, 3, 4, 5, 6, 7, 8, 9, 10, 11,
                                 12, 13, 14, 15);
}

// xor-lane shuffle via one ds_swizzle (group-of-32 mode: and=0x1f, xor=M)
template <int M>
__device__ __forceinline__ float swz(float x) {
  return __int_as_float(
      __builtin_amdgcn_ds_swizzle(__float_as_int(x), (M << 10) | 0x1f));
}
// batched reductions over the 16-lane half-wave: 8 independent xor-chains
// stepped together so the scheduler can issue 8 swizzles per wait.
__device__ __forceinline__ void rsum16x8(float* v) {
#pragma unroll
  for (int r = 0; r < 8; ++r) v[r] += swz<1>(v[r]);
#pragma unroll
  for (int r = 0; r < 8; ++r) v[r] += swz<2>(v[r]);
#pragma unroll
  for (int r = 0; r < 8; ++r) v[r] += swz<4>(v[r]);
#pragma unroll
  for (int r = 0; r < 8; ++r) v[r] += swz<8>(v[r]);
}
__device__ __forceinline__ void rmax16x8(float* v) {
#pragma unroll
  for (int r = 0; r < 8; ++r) v[r] = fmaxf(v[r], swz<1>(v[r]));
#pragma unroll
  for (int r = 0; r < 8; ++r) v[r] = fmaxf(v[r], swz<2>(v[r]));
#pragma unroll
  for (int r = 0; r < 8; ++r) v[r] = fmaxf(v[r], swz<4>(v[r]));
#pragma unroll
  for (int r = 0; r < 8; ++r) v[r] = fmaxf(v[r], swz<8>(v[r]));
}

// ---------------- weight fp32 -> f16 conversion ----------------
__global__ void cvt_kernel(const float* __restrict__ wq,
                           const float* __restrict__ wp,
                           _Float16* __restrict__ wqh,
                           _Float16* __restrict__ wph) {
  int i = blockIdx.x * 256 + threadIdx.x;
  if (i < 3 * CC * CC) wqh[i] = (_Float16)wq[i];
  if (i < CC * CC) wph[i] = (_Float16)wp[i];
}

// ---------------- relative position bias MLP ----------------
__global__ void bias_kernel(const float* __restrict__ w1,
                            const float* __restrict__ b1,
                            const float* __restrict__ w2,
                            const float* __restrict__ b2,
                            float* __restrict__ biasT) {
  int p = blockIdx.x * 256 + threadIdx.x;  // 0..4095 = i*64+j
  int i = p >> 6, j = p & 63;
  float d = (float)(i - j);
  float s = (d > 0.f) ? 1.f : ((d < 0.f) ? -1.f : 0.f);
  float rl = s * log1pf(fabsf(d));  // rel[:,0] == 0 always
  float acc[HH] = {};
  for (int n = 0; n < MH; ++n) {
    float hid = fmaxf(rl * w1[n * 2 + 1] + b1[n], 0.0f);
#pragma unroll
    for (int h = 0; h < HH; ++h) acc[h] += hid * w2[h * MH + n];
  }
#pragma unroll
  for (int h = 0; h < HH; ++h) biasT[(h << 12) + p] = acc[h] + b2[h];
}

// ---------------- fused per-window QKV + attention ----------------
__global__ __launch_bounds__(256) void win_attn_kernel(
    const float* __restrict__ x, const _Float16* __restrict__ wqkv,
    const float* __restrict__ bqkv, const float* __restrict__ lscale,
    const float* __restrict__ biasT, _Float16* __restrict__ aout) {
  __shared__ _Float16 sm[SM_HALVES];
  const int b = blockIdx.x;
  const int t = threadIdx.x;
  const int lane = t & 31;
  const int wu = __builtin_amdgcn_readfirstlane(t >> 5);  // wave id (scalar)
  const int nn = lane & 15;
  const int hf = lane >> 4;

  // ---- stage 1: x[b] (64x256 f32) -> LDS f16 (row stride 264) ----
  {
    const float4* xp = (const float4*)(x + (size_t)b * (LL * CC));
#pragma unroll
    for (int i = 0; i < 16; ++i) {
      int j = t + i * 256;
      int row = j >> 6;
      int c = (j & 63) << 2;
      float4 v = xp[j];
      v4h hv;
      hv[0] = (_Float16)v.x; hv[1] = (_Float16)v.y;
      hv[2] = (_Float16)v.z; hv[3] = (_Float16)v.w;
      *(v4h*)(sm + row * XS_STR + c) = hv;
    }
  }
  __syncthreads();

  // ---- stage 2: QKV GEMM (M=64, K=256), wave wu owns cols [96wu, 96wu+96) --
#pragma unroll 1
  for (int hb = 0; hb < 3; ++hb) {
    const int c3 = wu * 96 + hb * 32;  // 32-col head-block (scalar)
    const int which = c3 >> 8;         // 0=q 1=k 2=v (scalar)
    const int h = (c3 >> 5) & 7;       // head (scalar)
    const float bb0 = bqkv[c3 + nn];
    const float bb1 = bqkv[c3 + 16 + nn];
    v8f a0[4] = {}, a1[4] = {};
#pragma unroll 1
    for (int kt = 0; kt < 8; ++kt) {
      v16h A0 = tileA(sm, XS_STR, 0, kt * 32, lane);
      v16h A1 = tileA(sm, XS_STR, 16, kt * 32, lane);
      v16h A2 = tileA(sm, XS_STR, 32, kt * 32, lane);
      v16h A3 = tileA(sm, XS_STR, 48, kt * 32, lane);
      v16h B0 = tileB(wqkv, CC, c3, kt * 32, lane);
      v16h B1 = tileB(wqkv, CC, c3 + 16, kt * 32, lane);
      a0[0] = wmma_f16(A0, B0, a0[0]); a1[0] = wmma_f16(A0, B1, a1[0]);
      a0[1] = wmma_f16(A1, B0, a0[1]); a1[1] = wmma_f16(A1, B1, a1[1]);
      a0[2] = wmma_f16(A2, B0, a0[2]); a1[2] = wmma_f16(A2, B1, a1[2]);
      a0[3] = wmma_f16(A3, B0, a0[3]); a1[3] = wmma_f16(A3, B1, a1[3]);
    }
    // epilogue: bias, cosine-normalize q/k rows, scatter to per-head LDS
#pragma unroll
    for (int mt = 0; mt < 4; ++mt) {
      float v0[8], v1[8];
#pragma unroll
      for (int r = 0; r < 8; ++r) {
        v0[r] = a0[mt][r] + bb0;  // d = nn
        v1[r] = a1[mt][r] + bb1;  // d = nn+16
      }
      if (which < 2) {  // scalar branch
        float ss[8];
#pragma unroll
        for (int r = 0; r < 8; ++r) ss[r] = v0[r] * v0[r] + v1[r] * v1[r];
        rsum16x8(ss);
#pragma unroll
        for (int r = 0; r < 8; ++r) {
          // 1/max(sqrt(ss),1e-12) == rsq(max(ss,1e-24))
          float sc = __builtin_amdgcn_rsqf(fmaxf(ss[r], 1e-24f));
          v0[r] *= sc; v1[r] *= sc;
        }
      }
#pragma unroll
      for (int r = 0; r < 8; ++r) {
        int m = mt * 16 + r + hf * 8;
        if (which == 0) {
          _Float16* q = sm + QS_OFF + h * 2048 + m * 32;
          q[nn] = (_Float16)v0[r]; q[nn + 16] = (_Float16)v1[r];
        } else if (which == 1) {
          _Float16* k = sm + KS_OFF + h * 2048 + m * 32;
          k[nn] = (_Float16)v0[r]; k[nn + 16] = (_Float16)v1[r];
        } else {  // V stored transposed: vt[d][token]
          _Float16* vv = sm + VT_OFF + h * 2048;
          vv[nn * 64 + m] = (_Float16)v0[r];
          vv[(nn + 16) * 64 + m] = (_Float16)v1[r];
        }
      }
    }
  }
  __syncthreads();

  // ---- stage 3: attention, wave wu owns head h=wu ----
  {
    const int h = wu;
    const float scale = __expf(fminf(lscale[h], 4.6051702f));  // log(100)
    const _Float16* qh = sm + QS_OFF + h * 2048;
    const _Float16* kh = sm + KS_OFF + h * 2048;
    v16h AQ[4];
#pragma unroll
    for (int mt = 0; mt < 4; ++mt) AQ[mt] = tileA(qh, 32, mt * 16, 0, lane);
    v8f S[4][4] = {};
#pragma unroll
    for (int nt = 0; nt < 4; ++nt) {
      v16h BK = tileB(kh, 32, nt * 16, 0, lane);
#pragma unroll
      for (int mt = 0; mt < 4; ++mt) S[mt][nt] = wmma_f16(AQ[mt], BK, S[mt][nt]);
    }
    // scale + bias + row softmax fully in-register (row = half-wave),
    // reductions batched 8-wide per mt tile
    const float* bh = biasT + (h << 12);
#pragma unroll
    for (int mt = 0; mt < 4; ++mt) {
      float e0[8], e1[8], e2[8], e3[8], mx[8], sum[8];
#pragma unroll
      for (int r = 0; r < 8; ++r) {
        int row = mt * 16 + r + hf * 8;
        const float* br = bh + row * 64 + nn;
        e0[r] = S[mt][0][r] * scale + br[0];
        e1[r] = S[mt][1][r] * scale + br[16];
        e2[r] = S[mt][2][r] * scale + br[32];
        e3[r] = S[mt][3][r] * scale + br[48];
        mx[r] = fmaxf(fmaxf(e0[r], e1[r]), fmaxf(e2[r], e3[r]));
      }
      rmax16x8(mx);
#pragma unroll
      for (int r = 0; r < 8; ++r) {
        e0[r] = __expf(e0[r] - mx[r]); e1[r] = __expf(e1[r] - mx[r]);
        e2[r] = __expf(e2[r] - mx[r]); e3[r] = __expf(e3[r] - mx[r]);
        sum[r] = (e0[r] + e1[r]) + (e2[r] + e3[r]);
      }
      rsum16x8(sum);
#pragma unroll
      for (int r = 0; r < 8; ++r) {
        float inv = __builtin_amdgcn_rcpf(sum[r]);
        int row = mt * 16 + r + hf * 8;
        _Float16* pr = sm + PS_OFF + h * 4096 + row * 64;
        pr[nn]      = (_Float16)(e0[r] * inv);
        pr[nn + 16] = (_Float16)(e1[r] * inv);
        pr[nn + 32] = (_Float16)(e2[r] * inv);
        pr[nn + 48] = (_Float16)(e3[r] * inv);
      }
    }
    // O = P @ V  (P: 64x64 f16 in LDS, V^T as B operand)
    const _Float16* ph = sm + PS_OFF + h * 4096;
    const _Float16* vh = sm + VT_OFF + h * 2048;
    v8f O[4][2] = {};
#pragma unroll
    for (int kt = 0; kt < 2; ++kt) {
      v16h AP0 = tileA(ph, 64, 0, kt * 32, lane);
      v16h AP1 = tileA(ph, 64, 16, kt * 32, lane);
      v16h AP2 = tileA(ph, 64, 32, kt * 32, lane);
      v16h AP3 = tileA(ph, 64, 48, kt * 32, lane);
#pragma unroll
      for (int nt = 0; nt < 2; ++nt) {
        v16h BV = tileB(vh, 64, nt * 16, kt * 32, lane);
        O[0][nt] = wmma_f16(AP0, BV, O[0][nt]);
        O[1][nt] = wmma_f16(AP1, BV, O[1][nt]);
        O[2][nt] = wmma_f16(AP2, BV, O[2][nt]);
        O[3][nt] = wmma_f16(AP3, BV, O[3][nt]);
      }
    }
    // store O as f16 into (B, L, H*D) staging
    _Float16* op = aout + (size_t)b * (LL * CC) + h * DD;
#pragma unroll
    for (int mt = 0; mt < 4; ++mt)
#pragma unroll
      for (int r = 0; r < 8; ++r) {
        int row = mt * 16 + r + hf * 8;
        op[row * CC + nn] = (_Float16)O[mt][0][r];
        op[row * CC + 16 + nn] = (_Float16)O[mt][1][r];
      }
  }
}

// ---------------- output projection GEMM ----------------
__global__ __launch_bounds__(256) void proj_kernel(
    const _Float16* __restrict__ xin, const _Float16* __restrict__ wp,
    const float* __restrict__ bp, float* __restrict__ out) {
  __shared__ _Float16 xs[LL * XS_STR];
  const int t = threadIdx.x, lane = t & 31;
  const int wu = __builtin_amdgcn_readfirstlane(t >> 5);
  const int nn = lane & 15, hf = lane >> 4;
  const size_t r0 = (size_t)blockIdx.x * LL;
  {
    const v8h* xp = (const v8h*)(xin + r0 * CC);
#pragma unroll
    for (int i = 0; i < 8; ++i) {
      int j = t + i * 256;      // 0..2047 chunks of 8 halves
      int row = j >> 5;
      int c = (j & 31) << 3;
      *(v8h*)(xs + row * XS_STR + c) = xp[j];
    }
  }
  __syncthreads();
  const int n0 = wu * 32;
  const float b0 = bp[n0 + nn], b1 = bp[n0 + 16 + nn];
  v8f acc0[4] = {}, acc1[4] = {};
#pragma unroll 1
  for (int kt = 0; kt < 8; ++kt) {
    v16h A0 = tileA(xs, XS_STR, 0, kt * 32, lane);
    v16h A1 = tileA(xs, XS_STR, 16, kt * 32, lane);
    v16h A2 = tileA(xs, XS_STR, 32, kt * 32, lane);
    v16h A3 = tileA(xs, XS_STR, 48, kt * 32, lane);
    v16h B0 = tileB(wp, CC, n0, kt * 32, lane);
    v16h B1 = tileB(wp, CC, n0 + 16, kt * 32, lane);
    acc0[0] = wmma_f16(A0, B0, acc0[0]); acc1[0] = wmma_f16(A0, B1, acc1[0]);
    acc0[1] = wmma_f16(A1, B0, acc0[1]); acc1[1] = wmma_f16(A1, B1, acc1[1]);
    acc0[2] = wmma_f16(A2, B0, acc0[2]); acc1[2] = wmma_f16(A2, B1, acc1[2]);
    acc0[3] = wmma_f16(A3, B0, acc0[3]); acc1[3] = wmma_f16(A3, B1, acc1[3]);
  }
#pragma unroll
  for (int mt = 0; mt < 4; ++mt)
#pragma unroll
    for (int r = 0; r < 8; ++r) {
      int row = mt * 16 + r + hf * 8;
      out[(r0 + row) * CC + n0 + nn] = acc0[mt][r] + b0;
      out[(r0 + row) * CC + n0 + 16 + nn] = acc1[mt][r] + b1;
    }
}

extern "C" void kernel_launch(void* const* d_in, const int* in_sizes, int n_in,
                              void* d_out, int out_size, void* d_ws,
                              size_t ws_size, hipStream_t stream) {
  const float* x = (const float*)d_in[0];
  const float* qkvw = (const float*)d_in[1];
  const float* qkvb = (const float*)d_in[2];
  const float* projw = (const float*)d_in[3];
  const float* projb = (const float*)d_in[4];
  const float* ls = (const float*)d_in[5];
  const float* m1w = (const float*)d_in[6];
  const float* m1b = (const float*)d_in[7];
  const float* m2w = (const float*)d_in[8];
  const float* m2b = (const float*)d_in[9];

  // workspace layout (bytes): attn_out f16 (134217728) | qkv_w f16 (393216)
  //                           | proj_w f16 (131072) | bias fp32 (131072)
  uint8_t* ws = (uint8_t*)d_ws;
  _Float16* attn_out = (_Float16*)ws;
  _Float16* wq_h = (_Float16*)(ws + 134217728u);
  _Float16* wp_h = (_Float16*)(ws + 134217728u + 393216u);
  float* biasT = (float*)(ws + 134217728u + 393216u + 131072u);

  cvt_kernel<<<768, 256, 0, stream>>>(qkvw, projw, wq_h, wp_h);
  bias_kernel<<<16, 256, 0, stream>>>(m1w, m1b, m2w, m2b, biasT);
  win_attn_kernel<<<WB, 256, 0, stream>>>(x, wq_h, qkvb, ls, biasT, attn_out);
  proj_kernel<<<WB, 256, 0, stream>>>(attn_out, wp_h, projb, (float*)d_out);
}
